// Model_retirement_15015205667426
// MI455X (gfx1250) — matrix operand order repaired and checked by reading, not compiled, not run on hardware
//
#include <hip/hip_runtime.h>
#include <math.h>

#define TPB 256
#define WAVES 8

typedef __attribute__((ext_vector_type(16))) _Float16 v16h;
typedef __attribute__((ext_vector_type(8)))  float    v8f;

// ---------------- workspace layout (float offsets) ----------------
#define WS_THS  0      // theta column sums [48]
#define WS_THQ  48     // theta column sumsq [48]
#define WS_EDUS 96
#define WS_EDUQ 97
#define WS_AS   98     // entry-a stats per year, slots 0..61
#define WS_AQ   160
#define WS_YS   222    // y-column stats, slots 0..47
#define WS_YQ   270
#define WS_PRS  318    // entry-pr stats for years 40..47 (slot i-40)
#define WS_PRQ  326
#define WS_BENS 334    // entry-benefit stats for years 40..60 (slot i-40)
#define WS_BENQ 355
#define WS_NSTAT 384   // then: a_state[B], pr_state[B], ben_state[B], ysum_state[B],
                       // then: packed f16 weight fragments (48 years * 5 frags * 512 halfs)

// ---------------- d_out bases (units of B) ----------------
#define O_RWPR 0    // (B,8)
#define O_WC   8    // (B,40)
#define O_RWC  48   // (B,8,2)
#define O_RETC 64   // (B,13)
#define O_WA   77   // (B,40)
#define O_RWA  117  // (B,9,2)
#define O_RETA 135  // (B,14)
#define O_ALLH 149  // (B,61)
#define O_Y    210  // (B,48)

struct P {
  const float *theta,*edu,*a1,*wt,*bng,*bnb,*g1w,*g1b,*g2w,*g2b,
              *thw,*thb,*taw,*tab,*yhw,*yhb,*yaw,*yab,
              *pbng,*pbnb,*trw,*trb,*yrw,*yrb,
              *rbng,*rbnb,*rl1w,*rl1b,*rl2w,*rl2b;
  float *out;
  float *ws;
  int   Bn;
};

__device__ __forceinline__ float gelu_f(float x) {
  return 0.5f * x * (1.0f + erff(x * 0.70710678118654752f));
}
__device__ __forceinline__ float sigm(float x) {
  return 1.0f / (1.0f + expf(-x));
}

__device__ __forceinline__ _Float16* frag_base(const P& p) {
  return (_Float16*)(p.ws + WS_NSTAT + 4 * (size_t)p.Bn);
}

// One WMMA K-step: C += W_frag(16x32,f16) x X^T(32x16,f16).
// A operand pre-packed in fragment layout: frag[lane*16 + e] (one 32B global load).
// B operand (this code's contiguous k order, same as the k_pack A order): lane L holds N=L&15; element e -> K = e + 16*(L>>4)
// (contiguous in LDS row, one 32B LDS load).
__device__ __forceinline__ v8f wmma_step(const _Float16* __restrict__ frag,
                                         const _Float16* sxk, int lane, v8f c) {
  const int m  = lane & 15;
  const int hf = lane >> 4;
  v16h a = *(const v16h*)(frag + lane * 16);
  v16h b = *(const v16h*)(sxk + m * 64 + 16 * hf);     // note: caller bakes ldx into sxk/m stride
  return __builtin_amdgcn_wmma_f32_16x16x32_f16(false, a, false, b, (short)0, c, false, false);
}

// Same but for the 10x10 layers staged at ldx = 32 halfs.
__device__ __forceinline__ v8f wmma_step32(const _Float16* __restrict__ frag,
                                           const _Float16* sx, int lane, v8f c) {
  const int m  = lane & 15;
  const int hf = lane >> 4;
  v16h a = *(const v16h*)(frag + lane * 16);
  v16h b = *(const v16h*)(sx + m * 32 + 16 * hf);
  return __builtin_amdgcn_wmma_f32_16x16x32_f16(false, a, false, b, (short)0, c, false, false);
}

// D layout (32-bit C/D 16x16): lane L, VGPR v -> row M = v + 8*(L>>4) (out channel), col N = L&15 (batch row).
// Apply bias+gelu on channels <10, stage row-major f16 [16 rows x 32], zero-fill channels 10..31.
__device__ __forceinline__ void stage_act(v8f c, const float* __restrict__ bias, _Float16* sh, int lane) {
  const int r  = lane & 15;
  const int hf = lane >> 4;
#pragma unroll
  for (int v = 0; v < 8; ++v) {
    int ch = hf * 8 + v;
    float val = 0.0f;
    if (ch < 10) val = gelu_f(c[v] + bias[ch]);
    sh[r * 32 + ch] = (_Float16)val;
    sh[r * 32 + 16 + hf * 8 + v] = (_Float16)0.0f;
  }
}

// Shared trunk for years 0..47: BN coefficients -> stage features -> 4 WMMA layers -> heads.
__device__ __forceinline__ void trunk(const P& p, int i,
                                      _Float16* sx, _Float16* sh1, _Float16* sh2,
                                      _Float16* sh3, _Float16* sh4,
                                      float* a_ent_out, float* h_out, float* xx_out) {
  __shared__ float s_c0[64];
  __shared__ float s_c1[64];
  const int tid = threadIdx.x;
  const int w = tid >> 5, lane = tid & 31;
  const int b0 = (blockIdx.x * WAVES + w) * 16;
  const int Bn = p.Bn;
  const float invB = 1.0f / (float)Bn;
  const int d = (i == 0) ? 3 : 3 + i;
  const int ksteps = (d + 31) >> 5;           // 1 or 2
  float* as_state = p.ws + WS_NSTAT;

  // fold BN into val = raw*c0[f] + c1[f]; computed once per block
  if (tid < 64) {
    float c0 = 0.0f, c1 = 0.0f;
    if (tid < d) {
      float s, q;
      if (tid == 0)      { s = p.ws[WS_THS + i]; q = p.ws[WS_THQ + i]; }
      else if (tid == 1) { s = p.ws[WS_EDUS];    q = p.ws[WS_EDUQ]; }
      else if (tid == 2) { s = p.ws[WS_AS + i];  q = p.ws[WS_AQ + i]; }
      else               { s = p.ws[WS_YS + tid - 3]; q = p.ws[WS_YQ + tid - 3]; }
      float g  = p.bng[i * 50 + tid];
      float bb = p.bnb[i * 50 + tid];
      float mu  = s * invB;
      float var = q * invB - mu * mu;
      float rs  = rsqrtf(var + 1e-5f);
      c0 = g * rs;
      c1 = bb - g * mu * rs;
    }
    s_c0[tid] = c0; s_c1[tid] = c1;
  }
  __syncthreads();

  float a_ent = 0.0f;
  if (lane < 16) {                       // lanes 0-15 each stage one batch row
    const int b = b0 + lane;
    a_ent = (i == 0) ? p.a1[b] : as_state[b];
    _Float16* row = sx + lane * 64;
    float tv = p.theta[(long)b * 48 + i];
    float ev = p.edu[b];
    row[0] = (_Float16)(tv * s_c0[0] + s_c1[0]);
    row[1] = (_Float16)(ev * s_c0[1] + s_c1[1]);
    row[2] = (_Float16)(a_ent * s_c0[2] + s_c1[2]);
    const float* yrow = p.out + (long)O_Y * Bn + (long)b * 48;
    const int ycnt = d - 3;
    int j = 0;
    for (; j + 4 <= ycnt; j += 4) {       // 4 independent loads -> single wait
      float r0 = yrow[j], r1 = yrow[j + 1], r2 = yrow[j + 2], r3 = yrow[j + 3];
      row[3 + j] = (_Float16)(r0 * s_c0[3 + j] + s_c1[3 + j]);
      row[4 + j] = (_Float16)(r1 * s_c0[4 + j] + s_c1[4 + j]);
      row[5 + j] = (_Float16)(r2 * s_c0[5 + j] + s_c1[5 + j]);
      row[6 + j] = (_Float16)(r3 * s_c0[6 + j] + s_c1[6 + j]);
    }
    for (; j < ycnt; ++j)
      row[3 + j] = (_Float16)(yrow[j] * s_c0[3 + j] + s_c1[3 + j]);
    for (int f = d; f < 32 * ksteps; ++f) row[f] = (_Float16)0.0f;
  }
  __syncthreads();

  const _Float16* fb = frag_base(p) + (size_t)i * 5 * 512;
  v8f z = {};
  v8f c1v = wmma_step(fb, sx, lane, z);               // g1 k-step 0
  if (ksteps == 2) c1v = wmma_step(fb + 512, sx + 32, lane, c1v);  // g1 k-step 1
  stage_act(c1v, p.g1b + i * 10, sh1, lane);
  __syncthreads();
  v8f c2v = wmma_step32(fb + 2 * 512, sh1, lane, z);
  stage_act(c2v, p.g2b + i * 10, sh2, lane);
  __syncthreads();
  v8f c3v = wmma_step32(fb + 3 * 512, sh2, lane, z);
  stage_act(c3v, p.thb + i * 10, sh3, lane);
  v8f c4v = wmma_step32(fb + 4 * 512, sh2, lane, z);
  stage_act(c4v, p.tab + i * 10, sh4, lane);
  __syncthreads();

  float h = 0.0f, xx = 0.0f;
  if (lane < 16) {
    float lg[4];
#pragma unroll
    for (int o = 0; o < 4; ++o) {
      float acc = p.yhb[i * 4 + o];
#pragma unroll
      for (int k = 0; k < 10; ++k) acc += p.yhw[i * 40 + o * 10 + k] * (float)sh3[lane * 32 + k];
      lg[o] = acc;
    }
    float mx = fmaxf(fmaxf(lg[0], lg[1]), fmaxf(lg[2], lg[3]));
    float e0 = expf(lg[0] - mx), e1 = expf(lg[1] - mx), e2 = expf(lg[2] - mx), e3 = expf(lg[3] - mx);
    h = (1300.0f * e1 + 2080.0f * e2 + 2860.0f * e3) / (e0 + e1 + e2 + e3);
    float sa = p.yab[i];
#pragma unroll
    for (int k = 0; k < 10; ++k) sa += p.yaw[i * 10 + k] * (float)sh4[lane * 32 + k];
    xx = sigm(sa);
  }
  *a_ent_out = a_ent; *h_out = h; *xx_out = xx;
}

// ---------------- init kernels ----------------
__global__ void k_zero(P p) {
  for (int idx = threadIdx.x; idx < WS_NSTAT; idx += TPB) p.ws[idx] = 0.0f;
}

// Pack per-year layer weights into f16 A-fragment layout (zero-padded).
// frag ids per year: 0,1 = g1 k-steps; 2 = g2; 3 = th; 4 = ta. 512 halfs each.
__global__ __launch_bounds__(TPB) void k_pack(P p) {
  const int year = blockIdx.x / 5, frag = blockIdx.x % 5;
  _Float16* fb = frag_base(p) + (size_t)blockIdx.x * 512;
  const int d = (year == 0) ? 3 : 3 + year;
  const float* W; int ldw, Kreal, s;
  if (frag <= 1)      { W = p.g1w + year * 500; ldw = 50; Kreal = d;  s = frag; }
  else if (frag == 2) { W = p.g2w + year * 100; ldw = 10; Kreal = 10; s = 0; }
  else if (frag == 3) { W = p.thw + year * 100; ldw = 10; Kreal = 10; s = 0; }
  else                { W = p.taw + year * 100; ldw = 10; Kreal = 10; s = 0; }
  for (int idx = threadIdx.x; idx < 512; idx += TPB) {
    int lane = idx >> 4, e = idx & 15;
    int m = lane & 15, hf = lane >> 4;
    int k = 32 * s + e + 16 * hf;
    float wv = 0.0f;
    if (m < 10 && k < Kreal) wv = W[m * ldw + k];
    fb[idx] = (_Float16)wv;
  }
}

__global__ __launch_bounds__(TPB) void k_init(P p) {
  __shared__ float s[100];
  const int t = threadIdx.x;
  for (int idx = t; idx < 100; idx += TPB) s[idx] = 0.0f;
  __syncthreads();
  const int b = blockIdx.x * TPB + t;
  for (int jj = 0; jj < 48; ++jj) {
    int j = (jj + t) & 47;
    float v = p.theta[(long)b * 48 + j];
    atomicAdd(&s[j], v); atomicAdd(&s[48 + j], v * v);
  }
  float e = p.edu[b]; atomicAdd(&s[96], e); atomicAdd(&s[97], e * e);
  float a = p.a1[b];  atomicAdd(&s[98], a); atomicAdd(&s[99], a * a);
  for (int c = 48; c < 61; ++c) p.out[(long)O_ALLH * p.Bn + (long)b * 61 + c] = 0.0f; // all_h ret years
  __syncthreads();
  if (t < 48)       { atomicAdd(&p.ws[WS_THS + t], s[t]); atomicAdd(&p.ws[WS_THQ + t], s[48 + t]); }
  else if (t == 48) { atomicAdd(&p.ws[WS_EDUS], s[96]);   atomicAdd(&p.ws[WS_EDUQ], s[97]); }
  else if (t == 49) { atomicAdd(&p.ws[WS_AS],   s[98]);   atomicAdd(&p.ws[WS_AQ],   s[99]); }
}

// ---------------- working years i = 0..39 ----------------
__global__ __launch_bounds__(TPB) void k_work(P p, int i) {
  __shared__ __attribute__((aligned(32))) _Float16 s_x [WAVES][16 * 64];
  __shared__ __attribute__((aligned(32))) _Float16 s_h1[WAVES][16 * 32];
  __shared__ __attribute__((aligned(32))) _Float16 s_h2[WAVES][16 * 32];
  __shared__ __attribute__((aligned(32))) _Float16 s_h3[WAVES][16 * 32];
  __shared__ __attribute__((aligned(32))) _Float16 s_h4[WAVES][16 * 32];
  __shared__ float s_red[8];
  const int tid = threadIdx.x, w = tid >> 5, lane = tid & 31;
  if (tid < 8) s_red[tid] = 0.0f;

  float a_ent, h, xx;
  trunk(p, i, s_x[w], s_h1[w], s_h2[w], s_h3[w], s_h4[w], &a_ent, &h, &xx);

  const int Bn = p.Bn;
  float* as_state   = p.ws + WS_NSTAT;
  float* pr_state   = as_state + Bn;
  float* ben_state  = pr_state + Bn;
  float* ysum_state = ben_state + Bn;

  if (lane < 16) {
    const int b = (blockIdx.x * WAVES + w) * 16 + lane;
    float y   = p.wt[(long)b * 48 + i] * h;
    float net = y - 0.062f * y + a_ent;
    float cc  = xx * net + 1e-8f;
    float an  = (1.0f - xx) * net * 1.02f;
    p.out[(long)O_WC   * Bn + (long)b * 40 + i] = cc;
    p.out[(long)O_WA   * Bn + (long)b * 40 + i] = a_ent;
    p.out[(long)O_ALLH * Bn + (long)b * 61 + i] = h;
    p.out[(long)O_Y    * Bn + (long)b * 48 + i] = y;
    as_state[b] = an;
    float ysp = (i == 0) ? 0.0f : ysum_state[b];
    if (i == 39) {
      float ben = 0.4f * ysp * (1.0f / 35.0f);     // _benefit(ylist[:39], 0)
      ben_state[b] = ben;
      pr_state[b]  = 0.0f;
      p.out[(long)O_RWA * Bn + (long)b * 18 + 0] = an;  // rw_a[0] = (a_t, a_t)
      p.out[(long)O_RWA * Bn + (long)b * 18 + 1] = an;
      atomicAdd(&s_red[4], ben); atomicAdd(&s_red[5], ben * ben);
    }
    ysum_state[b] = ysp + y;
    atomicAdd(&s_red[0], an); atomicAdd(&s_red[1], an * an);
    atomicAdd(&s_red[2], y);  atomicAdd(&s_red[3], y * y);
  }
  __syncthreads();
  if (tid == 0) {
    atomicAdd(&p.ws[WS_AS + i + 1], s_red[0]); atomicAdd(&p.ws[WS_AQ + i + 1], s_red[1]);
    atomicAdd(&p.ws[WS_YS + i],     s_red[2]); atomicAdd(&p.ws[WS_YQ + i],     s_red[3]);
    if (i == 39) { atomicAdd(&p.ws[WS_BENS], s_red[4]); atomicAdd(&p.ws[WS_BENQ], s_red[5]); }
  }
}

// ---------------- retirement-window years i = 40..47 ----------------
__global__ __launch_bounds__(TPB) void k_rw(P p, int i) {
  __shared__ __attribute__((aligned(32))) _Float16 s_x [WAVES][16 * 64];
  __shared__ __attribute__((aligned(32))) _Float16 s_h1[WAVES][16 * 32];
  __shared__ __attribute__((aligned(32))) _Float16 s_h2[WAVES][16 * 32];
  __shared__ __attribute__((aligned(32))) _Float16 s_h3[WAVES][16 * 32];
  __shared__ __attribute__((aligned(32))) _Float16 s_h4[WAVES][16 * 32];
  __shared__ float s_red[8];
  const int tid = threadIdx.x, w = tid >> 5, lane = tid & 31;
  if (tid < 8) s_red[tid] = 0.0f;

  float a_ent, h, xx;
  trunk(p, i, s_x[w], s_h1[w], s_h2[w], s_h3[w], s_h4[w], &a_ent, &h, &xx);

  const int Bn = p.Bn;
  const float invB = 1.0f / (float)Bn;
  float* as_state   = p.ws + WS_NSTAT;
  float* pr_state   = as_state + Bn;
  float* ben_state  = pr_state + Bn;
  float* ysum_state = ben_state + Bn;
  const int k = i - 40;

  if (lane < 16) {
    const int b = (blockIdx.x * WAVES + w) * 16 + lane;
    // transition block: xc = [x (g2 out), bn(pr)]
    float pr_ent = pr_state[b];
    float pmu  = p.ws[WS_PRS + k] * invB;
    float pvar = p.ws[WS_PRQ + k] * invB - pmu * pmu;
    float xp   = p.pbng[k] * (pr_ent - pmu) * rsqrtf(pvar + 1e-5f) + p.pbnb[k];
    float xr[10];
#pragma unroll
    for (int o = 0; o < 10; ++o) {
      float acc = p.trb[k * 10 + o];
#pragma unroll
      for (int j2 = 0; j2 < 10; ++j2) acc += p.trw[k * 110 + o * 11 + j2] * (float)s_h2[w][lane * 32 + j2];
      acc += p.trw[k * 110 + o * 11 + 10] * xp;
      xr[o] = gelu_f(acc);
    }
    float sr = p.yrb[k];
#pragma unroll
    for (int j2 = 0; j2 < 10; ++j2) sr += p.yrw[k * 10 + j2] * xr[j2];
    float pr_cur = sigm(1000.0f * sr);
    float pr_new = (1.0f - pr_ent) * pr_cur + pr_ent;

    // retiree consumption block on [t, a, benefit]
    float ben_ent = ben_state[b];
    float amu = p.ws[WS_AS + i] * invB,   avar = p.ws[WS_AQ + i] * invB - amu * amu;
    float bmu = p.ws[WS_BENS + k] * invB, bvar = p.ws[WS_BENQ + k] * invB - bmu * bmu;
    float x0 = p.rbnb[k * 3];   // t feature: var=0 -> bn output = bias
    float x1 = p.rbng[k * 3 + 1] * (a_ent   - amu) * rsqrtf(avar + 1e-5f) + p.rbnb[k * 3 + 1];
    float x2 = p.rbng[k * 3 + 2] * (ben_ent - bmu) * rsqrtf(bvar + 1e-5f) + p.rbnb[k * 3 + 2];
    float h5[5];
#pragma unroll
    for (int o = 0; o < 5; ++o)
      h5[o] = gelu_f(p.rl1b[k * 5 + o] + p.rl1w[k * 15 + o * 3] * x0
                     + p.rl1w[k * 15 + o * 3 + 1] * x1 + p.rl1w[k * 15 + o * 3 + 2] * x2);
    float sv = p.rl2b[k];
#pragma unroll
    for (int o = 0; o < 5; ++o) sv += p.rl2w[k * 5 + o] * h5[o];
    float xtr = sigm(sv);

    // economics
    float y    = p.wt[(long)b * 48 + i] * h;
    float net  = y - 0.062f * y + a_ent;
    float cw   = xx * net + 1e-8f;
    float aw   = (1.0f - xx) * net * 1.02f;
    float cr   = xtr * (ben_ent + a_ent) + 1e-8f;
    float ar   = (1.0f - xtr) * (ben_ent + a_ent) * 1.02f;
    float an   = (1.0f - pr_new) * aw + pr_new * ar;
    float ysp  = ysum_state[b];
    float bcan = 0.4f * ysp * (1.0f / 35.0f) * (1.0f + 0.08f * (float)k);
    float bnx  = (1.0f - pr_new) * ben_ent + pr_new * bcan;
    float yst  = (1.0f - pr_new) * y;

    p.out[(long)O_RWPR * Bn + (long)b * 8  + k]               = pr_new;
    p.out[(long)O_RWC  * Bn + (long)b * 16 + k * 2]           = cw;
    p.out[(long)O_RWC  * Bn + (long)b * 16 + k * 2 + 1]       = cr;
    p.out[(long)O_RWA  * Bn + (long)b * 18 + (k + 1) * 2]     = aw;
    p.out[(long)O_RWA  * Bn + (long)b * 18 + (k + 1) * 2 + 1] = ar;
    p.out[(long)O_ALLH * Bn + (long)b * 61 + i]               = (1.0f - pr_new) * h;
    p.out[(long)O_Y    * Bn + (long)b * 48 + i]               = yst;
    if (i == 47) p.out[(long)O_RETA * Bn + (long)b * 14]      = an;   // ret_a[0]

    as_state[b] = an; pr_state[b] = pr_new; ben_state[b] = bnx; ysum_state[b] = ysp + yst;

    atomicAdd(&s_red[0], an);  atomicAdd(&s_red[1], an * an);
    atomicAdd(&s_red[2], yst); atomicAdd(&s_red[3], yst * yst);
    atomicAdd(&s_red[4], bnx); atomicAdd(&s_red[5], bnx * bnx);
    if (i < 47) { atomicAdd(&s_red[6], pr_new); atomicAdd(&s_red[7], pr_new * pr_new); }
  }
  __syncthreads();
  if (tid == 0) {
    atomicAdd(&p.ws[WS_AS + i + 1],   s_red[0]); atomicAdd(&p.ws[WS_AQ + i + 1],   s_red[1]);
    atomicAdd(&p.ws[WS_YS + i],       s_red[2]); atomicAdd(&p.ws[WS_YQ + i],       s_red[3]);
    atomicAdd(&p.ws[WS_BENS + k + 1], s_red[4]); atomicAdd(&p.ws[WS_BENQ + k + 1], s_red[5]);
    if (i < 47) { atomicAdd(&p.ws[WS_PRS + k + 1], s_red[6]); atomicAdd(&p.ws[WS_PRQ + k + 1], s_red[7]); }
  }
}

// ---------------- retired years i = 48..60 ----------------
__global__ __launch_bounds__(TPB) void k_ret(P p, int i) {
  __shared__ float s_red[4];
  const int tid = threadIdx.x;
  if (tid < 4) s_red[tid] = 0.0f;
  __syncthreads();
  const int Bn = p.Bn;
  const int b = blockIdx.x * TPB + tid;
  const int k = i - 40;
  const float invB = 1.0f / (float)Bn;
  float* as_state  = p.ws + WS_NSTAT;
  float* ben_state = as_state + 2 * Bn;

  float a_ent = as_state[b];
  float ben   = ben_state[b];
  float amu = p.ws[WS_AS + i] * invB,   avar = p.ws[WS_AQ + i] * invB - amu * amu;
  float bmu = p.ws[WS_BENS + k] * invB, bvar = p.ws[WS_BENQ + k] * invB - bmu * bmu;
  float x0 = p.rbnb[k * 3];
  float x1 = p.rbng[k * 3 + 1] * (a_ent - amu) * rsqrtf(avar + 1e-5f) + p.rbnb[k * 3 + 1];
  float x2 = p.rbng[k * 3 + 2] * (ben - bmu) * rsqrtf(bvar + 1e-5f) + p.rbnb[k * 3 + 2];
  float h5[5];
#pragma unroll
  for (int o = 0; o < 5; ++o)
    h5[o] = gelu_f(p.rl1b[k * 5 + o] + p.rl1w[k * 15 + o * 3] * x0
                   + p.rl1w[k * 15 + o * 3 + 1] * x1 + p.rl1w[k * 15 + o * 3 + 2] * x2);
  float sv = p.rl2b[k];
#pragma unroll
  for (int o = 0; o < 5; ++o) sv += p.rl2w[k * 5 + o] * h5[o];
  float xt = sigm(sv);
  float cand = ben + a_ent;
  float cc = xt * cand + 1e-8f;
  float an = (1.0f - xt) * cand * 1.02f;
  p.out[(long)O_RETC * Bn + (long)b * 13 + (i - 48)] = cc;
  p.out[(long)O_RETA * Bn + (long)b * 14 + (i - 47)] = an;
  as_state[b] = an;
  if (i < 60) {
    atomicAdd(&s_red[0], an);  atomicAdd(&s_red[1], an * an);
    atomicAdd(&s_red[2], ben); atomicAdd(&s_red[3], ben * ben);
    __syncthreads();
    if (tid == 0) {
      atomicAdd(&p.ws[WS_AS + i + 1],   s_red[0]); atomicAdd(&p.ws[WS_AQ + i + 1],   s_red[1]);
      atomicAdd(&p.ws[WS_BENS + k + 1], s_red[2]); atomicAdd(&p.ws[WS_BENQ + k + 1], s_red[3]);
    }
  }
}

// ---------------- host launcher ----------------
extern "C" void kernel_launch(void* const* d_in, const int* in_sizes, int n_in,
                              void* d_out, int out_size, void* d_ws, size_t ws_size,
                              hipStream_t stream) {
  P p;
  p.theta = (const float*)d_in[0];  p.edu  = (const float*)d_in[1];
  p.a1    = (const float*)d_in[2];  p.wt   = (const float*)d_in[3];
  p.bng   = (const float*)d_in[4];  p.bnb  = (const float*)d_in[5];
  p.g1w   = (const float*)d_in[6];  p.g1b  = (const float*)d_in[7];
  p.g2w   = (const float*)d_in[8];  p.g2b  = (const float*)d_in[9];
  p.thw   = (const float*)d_in[10]; p.thb  = (const float*)d_in[11];
  p.taw   = (const float*)d_in[12]; p.tab  = (const float*)d_in[13];
  p.yhw   = (const float*)d_in[14]; p.yhb  = (const float*)d_in[15];
  p.yaw   = (const float*)d_in[16]; p.yab  = (const float*)d_in[17];
  p.pbng  = (const float*)d_in[18]; p.pbnb = (const float*)d_in[19];
  p.trw   = (const float*)d_in[20]; p.trb  = (const float*)d_in[21];
  p.yrw   = (const float*)d_in[22]; p.yrb  = (const float*)d_in[23];
  p.rbng  = (const float*)d_in[24]; p.rbnb = (const float*)d_in[25];
  p.rl1w  = (const float*)d_in[26]; p.rl1b = (const float*)d_in[27];
  p.rl2w  = (const float*)d_in[28]; p.rl2b = (const float*)d_in[29];
  p.out = (float*)d_out;
  p.ws  = (float*)d_ws;
  const int Bn = in_sizes[1];     // batch size from edu
  p.Bn = Bn;

  k_zero<<<dim3(1), dim3(TPB), 0, stream>>>(p);
  k_pack<<<dim3(48 * 5), dim3(TPB), 0, stream>>>(p);
  k_init<<<dim3(Bn / TPB), dim3(TPB), 0, stream>>>(p);
  for (int i = 0; i < 40; ++i)  k_work<<<dim3(Bn / 128), dim3(TPB), 0, stream>>>(p, i);
  for (int i = 40; i < 48; ++i) k_rw<<<dim3(Bn / 128), dim3(TPB), 0, stream>>>(p, i);
  for (int i = 48; i < 61; ++i) k_ret<<<dim3(Bn / TPB), dim3(TPB), 0, stream>>>(p, i);
}
